// MoE_67242007986669
// MI455X (gfx1250) — matrix operand order repaired and checked by reading, not compiled, not run on hardware
//
#include <hip/hip_runtime.h>
#include <hip/hip_bf16.h>
#include <stdint.h>

// MoE (shared SwiGLU + 8 experts, top-2) for MI455X gfx1250.
// - all GEMMs: v_wmma_f32_16x16x32_bf16 (f32 accumulate)
// - LDS staging: GLOBAL_LOAD_ASYNC_TO_LDS_B128, double-buffered, ASYNCcnt-paced
// - weights converted f32->bf16 and pre-transposed so every copy is b128-contiguous

#define D_MODEL 1024
#define D_HID   2048
#define NEXP    8
#define NTOK    8192   // B*T = 4*2048

typedef __attribute__((ext_vector_type(16))) __bf16 v16bf;
typedef __attribute__((ext_vector_type(8)))  float  v8f;

union FragB { v16bf v; uint4 q[2]; };

__device__ __forceinline__ unsigned short f2bf(float f) {
  unsigned int u = __float_as_uint(f);
  u += 0x7FFFu + ((u >> 16) & 1u);              // round-to-nearest-even
  return (unsigned short)(u >> 16);
}

// 64-byte async copy: global -> LDS (4 x b128; INST_OFFSET applies to both sides)
__device__ __forceinline__ void async_cp64(unsigned lds, const void* g) {
  asm volatile(
      "global_load_async_to_lds_b128 %0, %1, off\n\t"
      "global_load_async_to_lds_b128 %0, %1, off offset:16\n\t"
      "global_load_async_to_lds_b128 %0, %1, off offset:32\n\t"
      "global_load_async_to_lds_b128 %0, %1, off offset:48"
      :: "v"(lds), "v"(g) : "memory");
}
// 32-byte async copy: global -> LDS (2 x b128)
__device__ __forceinline__ void async_cp32(unsigned lds, const void* g) {
  asm volatile(
      "global_load_async_to_lds_b128 %0, %1, off\n\t"
      "global_load_async_to_lds_b128 %0, %1, off offset:16"
      :: "v"(lds), "v"(g) : "memory");
}
#define ASYNC_WAIT(imm) asm volatile("s_wait_asynccnt " #imm ::: "memory")

__device__ __forceinline__ unsigned lds_off(const void* p) {
  return (unsigned)(uintptr_t)p;                // generic LDS addr: low 32 bits = DS offset
}

// ---------------- init: zero out, counts; identity list for "expert 8" (shared)
__global__ void k_init(float* __restrict__ out, int* __restrict__ counts,
                       int* __restrict__ idx8, float* __restrict__ w8) {
  const long long total = (long long)NTOK * D_MODEL;
  const long long stride = (long long)gridDim.x * blockDim.x;
  for (long long i = (long long)blockIdx.x * blockDim.x + threadIdx.x; i < total; i += stride) {
    out[i] = 0.f;
    if (i < NTOK) { idx8[i] = (int)i; w8[i] = 1.f; }
    if (i < 16)   counts[i] = (i == 8) ? NTOK : 0;
  }
}

// ---------------- f32 -> bf16 streaming convert (x only; row-major kept)
__global__ void k_cvt(const float* __restrict__ s, unsigned short* __restrict__ d, long long n) {
  const long long stride = (long long)gridDim.x * blockDim.x;
  for (long long i = (long long)blockIdx.x * blockDim.x + threadIdx.x; i < n; i += stride)
    d[i] = f2bf(s[i]);
}

// ---------------- f32 [mat][K][N] -> bf16 transposed [mat][N][K], tiled 32x32
__global__ void k_cvt_t(const float* __restrict__ src, unsigned short* __restrict__ dst,
                        int K, int N) {
  __shared__ unsigned short t[32][34];
  const size_t mo = (size_t)blockIdx.z * K * N;
  const int k0 = blockIdx.x * 32;
  const int n0 = blockIdx.y * 32;
  const int tid = threadIdx.x;
  const int r  = tid >> 3;
  const int c4 = (tid & 7) * 4;

  const float4 v = *(const float4*)&src[mo + (size_t)(k0 + r) * N + n0 + c4];
  t[r][c4 + 0] = f2bf(v.x);
  t[r][c4 + 1] = f2bf(v.y);
  t[r][c4 + 2] = f2bf(v.z);
  t[r][c4 + 3] = f2bf(v.w);
  __syncthreads();

  const unsigned int lo = (unsigned int)t[c4 + 0][r] | ((unsigned int)t[c4 + 1][r] << 16);
  const unsigned int hi = (unsigned int)t[c4 + 2][r] | ((unsigned int)t[c4 + 3][r] << 16);
  uint2 o; o.x = lo; o.y = hi;
  *(uint2*)&dst[mo + (size_t)(n0 + r) * K + k0 + c4] = o;
}

// ---------------- router: logits -> softmax -> top2 -> append to per-expert lists
__global__ void k_router(const float* __restrict__ x, const float* __restrict__ rw,
                         int* __restrict__ counts, int* __restrict__ idxl,
                         float* __restrict__ wl) {
  const int lane = threadIdx.x & 31;
  const int wave = threadIdx.x >> 5;
  const int token = blockIdx.x * 8 + wave;
  if (token >= NTOK) return;

  float acc[NEXP] = {0.f, 0.f, 0.f, 0.f, 0.f, 0.f, 0.f, 0.f};
  for (int d = lane; d < D_MODEL; d += 32) {
    const float xv = x[(size_t)token * D_MODEL + d];
    const float4* rp = (const float4*)&rw[d * NEXP];
    const float4 r0 = rp[0], r1 = rp[1];
    acc[0] += xv * r0.x; acc[1] += xv * r0.y; acc[2] += xv * r0.z; acc[3] += xv * r0.w;
    acc[4] += xv * r1.x; acc[5] += xv * r1.y; acc[6] += xv * r1.z; acc[7] += xv * r1.w;
  }
#pragma unroll
  for (int m = 16; m >= 1; m >>= 1)
#pragma unroll
    for (int e = 0; e < NEXP; ++e) acc[e] += __shfl_xor(acc[e], m, 32);

  if (lane == 0) {
    float mx = acc[0];
#pragma unroll
    for (int e = 1; e < NEXP; ++e) mx = fmaxf(mx, acc[e]);
    float p[NEXP], s = 0.f;
#pragma unroll
    for (int e = 0; e < NEXP; ++e) { p[e] = __expf(acc[e] - mx); s += p[e]; }
    const float inv = __builtin_amdgcn_rcpf(s);
#pragma unroll
    for (int e = 0; e < NEXP; ++e) p[e] *= inv;
    int b1 = 0;
#pragma unroll
    for (int e = 1; e < NEXP; ++e) if (p[e] > p[b1]) b1 = e;
    int b2 = (b1 == 0) ? 1 : 0;
#pragma unroll
    for (int e = 0; e < NEXP; ++e) if (e != b1 && p[e] > p[b2]) b2 = e;

    int pos = atomicAdd(&counts[b1], 1);
    idxl[b1 * NTOK + pos] = token; wl[b1 * NTOK + pos] = p[b1];
    pos = atomicAdd(&counts[b2], 1);
    idxl[b2 * NTOK + pos] = token; wl[b2 * NTOK + pos] = p[b2];
  }
}

// ---------------- gate/up GEMM + SiLU*up*routeWeight -> H (bf16)
// Block tile M=128 x N=64, K-step 64; async double-buffered LDS staging.
__global__ void __launch_bounds__(256, 2)
k_gateup(const unsigned short* __restrict__ xb,
         const unsigned short* __restrict__ gatebt,
         const unsigned short* __restrict__ upbt,
         const unsigned short* __restrict__ sgbt,
         const unsigned short* __restrict__ subt,
         const int* __restrict__ counts,
         const int* __restrict__ idxl,
         const float* __restrict__ wl,
         unsigned short* __restrict__ Hbuf) {
  __shared__ unsigned short lA[2][128 * 64];
  __shared__ unsigned short lBg[2][64 * 64];   // [n][k]
  __shared__ unsigned short lBu[2][64 * 64];
  __shared__ int   lTok[128];
  __shared__ float lWgt[128];

  const int e = blockIdx.z;
  const int count = counts[e];
  const int mtiles = (count + 127) >> 7;
  if ((int)blockIdx.x >= mtiles) return;

  const unsigned short* gw = (e < NEXP) ? gatebt + (size_t)e * D_MODEL * D_HID : sgbt;
  const unsigned short* uw = (e < NEXP) ? upbt   + (size_t)e * D_MODEL * D_HID : subt;

  const int mbase = blockIdx.x * 128;
  const int n0    = blockIdx.y * 64;
  const int tid   = threadIdx.x;
  const int lane  = tid & 31;
  const int wave  = tid >> 5;
  const int wm    = wave & 3;
  const int wn    = wave >> 2;

  if (tid < 128) {
    const int gr = mbase + tid;
    lTok[tid] = (gr < count) ? idxl[e * NTOK + gr] : 0;
    lWgt[tid] = (gr < count) ? wl[e * NTOK + gr] : 0.f;
  }
  __syncthreads();

  // per-thread staging assignment (loop-invariant bases)
  const int rA    = tid & 127;           // A row
  const int halfA = tid >> 7;            // A 32-elem K half
  const int nB    = tid & 63;            // B col
  const int kcB   = (tid >> 6) * 16;     // B 16-elem K chunk
  const int tkn   = lTok[rA];

  const unsigned short* gA = xb + (size_t)tkn * D_MODEL + halfA * 32;
  const unsigned short* gG = gw + (size_t)(n0 + nB) * D_MODEL + kcB;
  const unsigned short* gU = uw + (size_t)(n0 + nB) * D_MODEL + kcB;
  const unsigned ldsA[2]  = { lds_off(&lA[0][rA * 64 + halfA * 32]),
                              lds_off(&lA[1][rA * 64 + halfA * 32]) };
  const unsigned ldsBg[2] = { lds_off(&lBg[0][nB * 64 + kcB]),
                              lds_off(&lBg[1][nB * 64 + kcB]) };
  const unsigned ldsBu[2] = { lds_off(&lBu[0][nB * 64 + kcB]),
                              lds_off(&lBu[1][nB * 64 + kcB]) };

  v8f accg[2][2], accu[2][2];
#pragma unroll
  for (int i = 0; i < 2; ++i)
#pragma unroll
    for (int j = 0; j < 2; ++j) {
      accg[i][j] = (v8f){0.f,0.f,0.f,0.f,0.f,0.f,0.f,0.f};
      accu[i][j] = (v8f){0.f,0.f,0.f,0.f,0.f,0.f,0.f,0.f};
    }

  const int NK = D_MODEL / 64;
  // prologue: issue K-groups 0 and 1 (8 async ops each per wave)
  async_cp64(ldsA[0], gA);          async_cp64(ldsA[1], gA + 64);
  async_cp32(ldsBg[0], gG);         async_cp32(ldsBg[1], gG + 64);
  async_cp32(ldsBu[0], gU);         async_cp32(ldsBu[1], gU + 64);

  const int kbaseA = (lane & 16) ? 8 : 0;
  const int kbaseB = (lane & 16) ? 8 : 0;
  const int lidx   = lane & 15;

  for (int kk = 0; kk < NK; ++kk) {
    const int buf = kk & 1;
    if (kk == NK - 1) { ASYNC_WAIT(0x0); } else { ASYNC_WAIT(0x8); }
    __syncthreads();                 // buffer `buf` ready for all waves

#pragma unroll
    for (int kc = 0; kc < 2; ++kc) {
      FragB af[2], bg[2], bu[2];
#pragma unroll
      for (int i = 0; i < 2; ++i) {
        const int rowA = wm * 32 + i * 16 + lidx;
        af[i].q[0] = *(const uint4*)&lA[buf][rowA * 64 + kc * 32 + kbaseA];
        af[i].q[1] = *(const uint4*)&lA[buf][rowA * 64 + kc * 32 + kbaseA + 16];
      }
#pragma unroll
      for (int j = 0; j < 2; ++j) {
        const int colB = wn * 32 + j * 16 + lidx;
        bg[j].q[0] = *(const uint4*)&lBg[buf][colB * 64 + kc * 32 + kbaseB];
        bg[j].q[1] = *(const uint4*)&lBg[buf][colB * 64 + kc * 32 + kbaseB + 16];
        bu[j].q[0] = *(const uint4*)&lBu[buf][colB * 64 + kc * 32 + kbaseB];
        bu[j].q[1] = *(const uint4*)&lBu[buf][colB * 64 + kc * 32 + kbaseB + 16];
      }
#pragma unroll
      for (int i = 0; i < 2; ++i)
#pragma unroll
        for (int j = 0; j < 2; ++j) {
          accg[i][j] = __builtin_amdgcn_wmma_f32_16x16x32_bf16(
              false, af[i].v, false, bg[j].v, (short)0, accg[i][j], false, false);
          accu[i][j] = __builtin_amdgcn_wmma_f32_16x16x32_bf16(
              false, af[i].v, false, bu[j].v, (short)0, accu[i][j], false, false);
        }
    }
    __syncthreads();                 // everyone done reading buffer `buf`
    if (kk + 2 < NK) {               // refill it for K-group kk+2
      const int k0n = (kk + 2) * 64;
      async_cp64(ldsA[buf], gA + k0n);
      async_cp32(ldsBg[buf], gG + k0n);
      async_cp32(ldsBu[buf], gU + k0n);
    }
  }

  // Epilogue: h = silu(g) * u * route_weight  (fast: g * v_rcp(1+exp(-g)))
  const int moff = (lane & 16) ? 8 : 0;
#pragma unroll
  for (int i = 0; i < 2; ++i)
#pragma unroll
    for (int j = 0; j < 2; ++j) {
      const int col = n0 + wn * 32 + j * 16 + lidx;
#pragma unroll
      for (int el = 0; el < 8; ++el) {
        const int row = wm * 32 + i * 16 + moff + el;
        const int grow = mbase + row;
        if (grow < count) {
          const float g = accg[i][j][el];
          const float u = accu[i][j][el];
          const float sig = __builtin_amdgcn_rcpf(1.f + __expf(-g));
          const float h = g * sig * u * lWgt[row];
          Hbuf[((size_t)e * NTOK + grow) * D_HID + col] = f2bf(h);
        }
      }
    }
}

// ---------------- down GEMM: H[e] (M x 2048) @ down_w^T tiles, scatter-add
__global__ void __launch_bounds__(256, 2)
k_down(const unsigned short* __restrict__ Hbuf,
       const unsigned short* __restrict__ downbt,
       const unsigned short* __restrict__ sdbt,
       const int* __restrict__ counts,
       const int* __restrict__ idxl,
       float* __restrict__ out) {
  __shared__ unsigned short lA[2][128 * 64];
  __shared__ unsigned short lB[2][64 * 64];
  __shared__ int lTok[128];

  const int e = blockIdx.z;
  const int count = counts[e];
  const int mtiles = (count + 127) >> 7;
  if ((int)blockIdx.x >= mtiles) return;

  const unsigned short* dw = (e < NEXP) ? downbt + (size_t)e * D_HID * D_MODEL : sdbt;

  const int mbase = blockIdx.x * 128;
  const int n0    = blockIdx.y * 64;
  const int tid   = threadIdx.x;
  const int lane  = tid & 31;
  const int wave  = tid >> 5;
  const int wm    = wave & 3;
  const int wn    = wave >> 2;

  if (tid < 128) {
    const int gr = mbase + tid;
    lTok[tid] = (gr < count) ? idxl[e * NTOK + gr] : 0;
  }
  __syncthreads();

  const int rA    = tid & 127;
  const int halfA = tid >> 7;
  const int nB    = tid & 63;
  const int kcB   = (tid >> 6) * 16;

  const unsigned short* gA = Hbuf + ((size_t)e * NTOK + mbase + rA) * D_HID + halfA * 32;
  const unsigned short* gB = dw + (size_t)(n0 + nB) * D_HID + kcB;
  const unsigned ldsA[2] = { lds_off(&lA[0][rA * 64 + halfA * 32]),
                             lds_off(&lA[1][rA * 64 + halfA * 32]) };
  const unsigned ldsB[2] = { lds_off(&lB[0][nB * 64 + kcB]),
                             lds_off(&lB[1][nB * 64 + kcB]) };

  v8f acc[2][2];
#pragma unroll
  for (int i = 0; i < 2; ++i)
#pragma unroll
    for (int j = 0; j < 2; ++j)
      acc[i][j] = (v8f){0.f,0.f,0.f,0.f,0.f,0.f,0.f,0.f};

  const int NK = D_HID / 64;
  async_cp64(ldsA[0], gA);   async_cp64(ldsA[1], gA + 64);
  async_cp32(ldsB[0], gB);   async_cp32(ldsB[1], gB + 64);

  const int kbaseA = (lane & 16) ? 8 : 0;
  const int kbaseB = (lane & 16) ? 8 : 0;
  const int lidx   = lane & 15;

  for (int kk = 0; kk < NK; ++kk) {
    const int buf = kk & 1;
    if (kk == NK - 1) { ASYNC_WAIT(0x0); } else { ASYNC_WAIT(0x6); }
    __syncthreads();

#pragma unroll
    for (int kc = 0; kc < 2; ++kc) {
      FragB af[2], bf[2];
#pragma unroll
      for (int i = 0; i < 2; ++i) {
        const int rowA = wm * 32 + i * 16 + lidx;
        af[i].q[0] = *(const uint4*)&lA[buf][rowA * 64 + kc * 32 + kbaseA];
        af[i].q[1] = *(const uint4*)&lA[buf][rowA * 64 + kc * 32 + kbaseA + 16];
      }
#pragma unroll
      for (int j = 0; j < 2; ++j) {
        const int colB = wn * 32 + j * 16 + lidx;
        bf[j].q[0] = *(const uint4*)&lB[buf][colB * 64 + kc * 32 + kbaseB];
        bf[j].q[1] = *(const uint4*)&lB[buf][colB * 64 + kc * 32 + kbaseB + 16];
      }
#pragma unroll
      for (int i = 0; i < 2; ++i)
#pragma unroll
        for (int j = 0; j < 2; ++j)
          acc[i][j] = __builtin_amdgcn_wmma_f32_16x16x32_bf16(
              false, af[i].v, false, bf[j].v, (short)0, acc[i][j], false, false);
    }
    __syncthreads();
    if (kk + 2 < NK) {
      const int k0n = (kk + 2) * 64;
      async_cp64(ldsA[buf], gA + k0n);
      async_cp32(ldsB[buf], gB + k0n);
    }
  }

  const int moff = (lane & 16) ? 8 : 0;
#pragma unroll
  for (int i = 0; i < 2; ++i)
#pragma unroll
    for (int j = 0; j < 2; ++j) {
      const int col = n0 + wn * 32 + j * 16 + lidx;
#pragma unroll
      for (int el = 0; el < 8; ++el) {
        const int row = wm * 32 + i * 16 + moff + el;
        const int grow = mbase + row;
        if (grow < count) {
          const int t = lTok[row];
          unsafeAtomicAdd(&out[(size_t)t * D_MODEL + col], acc[i][j][el]);
        }
      }
    }
}

extern "C" void kernel_launch(void* const* d_in, const int* in_sizes, int n_in,
                              void* d_out, int out_size, void* d_ws, size_t ws_size,
                              hipStream_t stream) {
  const float* x  = (const float*)d_in[0];
  const float* rw = (const float*)d_in[1];
  const float* sg = (const float*)d_in[2];
  const float* su = (const float*)d_in[3];
  const float* sd = (const float*)d_in[4];
  const float* gw = (const float*)d_in[5];
  const float* uw = (const float*)d_in[6];
  const float* dw = (const float*)d_in[7];
  float* out = (float*)d_out;

  char* ws = (char*)d_ws;
  size_t off = 0;
  auto alloc = [&](size_t bytes) -> size_t {
    size_t o = off; off += (bytes + 255) & ~(size_t)255; return o;
  };
  unsigned short* xb   = (unsigned short*)(ws + alloc((size_t)NTOK * D_MODEL * 2));
  unsigned short* sgbt = (unsigned short*)(ws + alloc((size_t)D_MODEL * D_HID * 2));
  unsigned short* subt = (unsigned short*)(ws + alloc((size_t)D_MODEL * D_HID * 2));
  unsigned short* sdbt = (unsigned short*)(ws + alloc((size_t)D_HID * D_MODEL * 2));
  unsigned short* gbt  = (unsigned short*)(ws + alloc((size_t)NEXP * D_MODEL * D_HID * 2));
  unsigned short* ubt  = (unsigned short*)(ws + alloc((size_t)NEXP * D_MODEL * D_HID * 2));
  unsigned short* dbt  = (unsigned short*)(ws + alloc((size_t)NEXP * D_HID * D_MODEL * 2));
  unsigned short* Hbuf = (unsigned short*)(ws + alloc((size_t)9 * NTOK * D_HID * 2));
  int*   counts = (int*)(ws + alloc(64));
  int*   idxl   = (int*)(ws + alloc((size_t)9 * NTOK * 4));
  float* wl     = (float*)(ws + alloc((size_t)9 * NTOK * 4));

  k_init<<<4096, 256, 0, stream>>>(out, counts, idxl + 8 * NTOK, wl + 8 * NTOK);
  k_cvt<<<2048, 256, 0, stream>>>(x, xb, (long long)NTOK * D_MODEL);
  k_cvt_t<<<dim3(D_MODEL / 32, D_HID / 32, 1), 256, 0, stream>>>(sg, sgbt, D_MODEL, D_HID);
  k_cvt_t<<<dim3(D_MODEL / 32, D_HID / 32, 1), 256, 0, stream>>>(su, subt, D_MODEL, D_HID);
  k_cvt_t<<<dim3(D_HID / 32, D_MODEL / 32, 1), 256, 0, stream>>>(sd, sdbt, D_HID, D_MODEL);
  k_cvt_t<<<dim3(D_MODEL / 32, D_HID / 32, NEXP), 256, 0, stream>>>(gw, gbt, D_MODEL, D_HID);
  k_cvt_t<<<dim3(D_MODEL / 32, D_HID / 32, NEXP), 256, 0, stream>>>(uw, ubt, D_MODEL, D_HID);
  k_cvt_t<<<dim3(D_HID / 32, D_MODEL / 32, NEXP), 256, 0, stream>>>(dw, dbt, D_HID, D_MODEL);
  k_router<<<NTOK / 8, 256, 0, stream>>>(x, rw, counts, idxl, wl);
  k_gateup<<<dim3(64, D_HID / 64, 9), 256, 0, stream>>>(xb, gbt, ubt, sgbt, subt, counts,
                                                        idxl, wl, Hbuf);
  k_down<<<dim3(64, D_MODEL / 64, 9), 256, 0, stream>>>(Hbuf, dbt, sdbt, counts, idxl, out);
}
